// GraphAttentionLayer_36240934043864
// MI455X (gfx1250) — matrix operand order repaired and checked by reading, not compiled, not run on hardware
//
#include <hip/hip_runtime.h>
#include <hip/hip_bf16.h>

#define F 64
#define NHEADS 4

typedef __attribute__((ext_vector_type(16))) __bf16 v16bf;
typedef __attribute__((ext_vector_type(8)))  float  v8f;

// float atomic-max via signed/unsigned integer atomics (valid with -inf init).
__device__ __forceinline__ void atomicMaxFloat(float* addr, float val) {
    if (val >= 0.0f) atomicMax((int*)addr, __float_as_int(val));
    else             atomicMin((unsigned int*)addr, __float_as_uint(val));
}

__device__ __forceinline__ void cvt8(v16bf& d, int base, float4 x, float4 y) {
    d[base + 0] = (__bf16)x.x; d[base + 1] = (__bf16)x.y;
    d[base + 2] = (__bf16)x.z; d[base + 3] = (__bf16)x.w;
    d[base + 4] = (__bf16)y.x; d[base + 5] = (__bf16)y.y;
    d[base + 6] = (__bf16)y.z; d[base + 7] = (__bf16)y.w;
}

// ---------------------------------------------------------------------------
// One-time weight pack: Wt[c*64+k] = bf16(W[k*64+c]) (column-major bf16).
// Makes every WMMA B operand a 32-byte contiguous load.
// ---------------------------------------------------------------------------
__global__ void gat_pack_w(const float* __restrict__ Wk, const float* __restrict__ Wq,
                           const float* __restrict__ W,
                           __bf16* __restrict__ WkT, __bf16* __restrict__ WqT,
                           __bf16* __restrict__ WT) {
    int idx = blockIdx.x * blockDim.x + threadIdx.x;
    if (idx >= F * F) return;
    const int c = idx >> 6, k = idx & 63;
    WkT[idx] = (__bf16)Wk[(size_t)k * F + c];
    WqT[idx] = (__bf16)Wq[(size_t)k * F + c];
    WT[idx]  = (__bf16)W[(size_t)k * F + c];
}

// ---------------------------------------------------------------------------
// out[M,64] = bf16(A[M,64]) @ Wt (+ bias) via v_wmma_f32_16x16x32_bf16.
// A: float4 vector loads + cvt_pk packing in this kernel's contiguous K order (same as B)
//    (lane kh half: elements [0..15] = K kh*16+{0..15}).
// B: direct 32-byte v16bf loads from the pre-packed column-major bf16 weights
//    (lane kh half holds K = kh*16+{0..15} of each 32-deep K tile).
// Grid = M/32 blocks of 64 threads (2 waves); M % 32 == 0, no divergent guards
// so EXEC is all-ones at every WMMA.
// ---------------------------------------------------------------------------
__global__ __launch_bounds__(64) void gat_gemm64_wmma(
        const float* __restrict__ A, const __bf16* __restrict__ Wt,
        const float* __restrict__ bias, float* __restrict__ out, int M) {
    const int wave = blockIdx.x * 2 + (threadIdx.x >> 5);
    const int lane = threadIdx.x & 31;
    const int row0 = wave * 16;
    const int n    = lane & 15;   // A row within tile / B,C,D column
    const int kh   = lane >> 4;   // K-half select

    // A tile rows: two 32-deep K tiles, each = two contiguous float4 pairs.
    const float* arow = A + (size_t)(row0 + n) * F;
    float4 f0 = *(const float4*)(arow + kh * 16);
    float4 f1 = *(const float4*)(arow + kh * 16 + 4);
    float4 f2 = *(const float4*)(arow + kh * 16 + 8);
    float4 f3 = *(const float4*)(arow + kh * 16 + 12);
    float4 f4 = *(const float4*)(arow + 32 + kh * 16);
    float4 f5 = *(const float4*)(arow + 32 + kh * 16 + 4);
    float4 f6 = *(const float4*)(arow + 32 + kh * 16 + 8);
    float4 f7 = *(const float4*)(arow + 32 + kh * 16 + 12);
    v16bf a0, a1;
    cvt8(a0, 0, f0, f1); cvt8(a0, 8, f2, f3);
    cvt8(a1, 0, f4, f5); cvt8(a1, 8, f6, f7);

#pragma unroll
    for (int nt = 0; nt < 4; ++nt) {
        const int col = nt * 16 + n;
        const __bf16* wcol = Wt + (size_t)col * F;
        v16bf b0 = *(const v16bf*)(wcol + kh * 16);        // K =      kh*16 + 0..15
        v16bf b1 = *(const v16bf*)(wcol + 32 + kh * 16);   // K = 32 + kh*16 + 0..15

        v8f c = {};
        c = __builtin_amdgcn_wmma_f32_16x16x32_bf16(false, a0, false, b0, (short)0, c, false, false);
        c = __builtin_amdgcn_wmma_f32_16x16x32_bf16(false, a1, false, b1, (short)0, c, false, false);

        const float bv = bias ? bias[nt * 16 + n] : 0.0f;
        // C/D layout: M = kh*8 + v, N = n.
#pragma unroll
        for (int v = 0; v < 8; ++v)
            out[(size_t)(row0 + (kh << 3) + v) * F + nt * 16 + n] = c[v] + bv;
    }
}

// ---------------------------------------------------------------------------
// Init: m = -inf, z = 0, hn = 0 (re-run every call; harness does not re-zero ws)
// ---------------------------------------------------------------------------
__global__ void gat_init(float* __restrict__ mbuf, float* __restrict__ zbuf,
                         float* __restrict__ hn, int NH, long long hnElems) {
    long long i = (long long)blockIdx.x * blockDim.x + threadIdx.x;
    if (i < NH) { mbuf[i] = -__builtin_huge_valf(); zbuf[i] = 0.0f; }
    if (i < hnElems) hn[i] = 0.0f;
}

// ---------------------------------------------------------------------------
// Edge scores: e[e,h] = <k[src[e],h,:], q[dst[e],h,:]> + segment-max into mbuf.
// One thread per (edge, head); 16-float dots via float4 loads (L2-resident).
// ---------------------------------------------------------------------------
__global__ void gat_edge_score(const float* __restrict__ k, const float* __restrict__ q,
                               const int* __restrict__ src, const int* __restrict__ dst,
                               float* __restrict__ e, float* __restrict__ mbuf, int E) {
    int idx = blockIdx.x * blockDim.x + threadIdx.x;
    if (idx >= E * NHEADS) return;
    const int edge = idx >> 2, head = idx & 3;
    const int s = src[edge], d = dst[edge];
    const float4* kr = (const float4*)(k + (size_t)s * F + head * 16);
    const float4* qr = (const float4*)(q + (size_t)d * F + head * 16);
    float acc = 0.0f;
#pragma unroll
    for (int i = 0; i < 4; ++i) {
        float4 kv = kr[i], qv = qr[i];
        acc += kv.x * qv.x + kv.y * qv.y + kv.z * qv.z + kv.w * qv.w;
    }
    e[idx] = acc;
    atomicMaxFloat(&mbuf[(size_t)d * NHEADS + head], acc);
}

// ---------------------------------------------------------------------------
// ex = exp(e - m[dst]) (in place) + segment-sum into z.
// ---------------------------------------------------------------------------
__global__ void gat_edge_exp(const float* __restrict__ mbuf, const int* __restrict__ dst,
                             float* __restrict__ e, float* __restrict__ zbuf, int E) {
    int idx = blockIdx.x * blockDim.x + threadIdx.x;
    if (idx >= E * NHEADS) return;
    const int edge = idx >> 2, head = idx & 3;
    const int d = dst[edge];
    const float ex = __expf(e[idx] - mbuf[(size_t)d * NHEADS + head]);
    e[idx] = ex;
    atomicAdd(&zbuf[(size_t)d * NHEADS + head], ex);
}

// ---------------------------------------------------------------------------
// Aggregate: hn[dst,h,f] += (ex[e,h]/z[dst,h]) * h[src,f].
// 64 threads per edge (one per f); 4 L2 atomics each. hn (102MB) fits in L2.
// ---------------------------------------------------------------------------
__global__ void gat_aggregate(const float* __restrict__ h, const float* __restrict__ e,
                              const float* __restrict__ zbuf,
                              const int* __restrict__ src, const int* __restrict__ dst,
                              float* __restrict__ hn, int E) {
    int idx = blockIdx.x * blockDim.x + threadIdx.x;
    const int edge = idx >> 6, f = idx & 63;
    if (edge >= E) return;
    const int s = src[edge], d = dst[edge];
    const float hv = h[(size_t)s * F + f];
#pragma unroll
    for (int head = 0; head < NHEADS; ++head) {
        const float a = e[(size_t)edge * NHEADS + head] / zbuf[(size_t)d * NHEADS + head];
        atomicAdd(&hn[((size_t)d * NHEADS + head) * F + f], a * hv);
    }
}

extern "C" void kernel_launch(void* const* d_in, const int* in_sizes, int n_in,
                              void* d_out, int out_size, void* d_ws, size_t ws_size,
                              hipStream_t stream) {
    const float* h   = (const float*)d_in[0];
    const int*   src = (const int*)d_in[1];
    const int*   dst = (const int*)d_in[2];
    const float* Wk  = (const float*)d_in[3];
    const float* Wq  = (const float*)d_in[4];
    const float* W   = (const float*)d_in[5];
    const float* b   = (const float*)d_in[6];
    float* out = (float*)d_out;

    const int N = in_sizes[0] / F;   // 100000
    const int E = in_sizes[1];       // 1600000

    // Workspace carve-up: packed bf16 weights first (32B-aligned tiles), then fp32.
    __bf16* WkT = (__bf16*)d_ws;                 // 64*64 bf16
    __bf16* WqT = WkT + F * F;
    __bf16* WT  = WqT + F * F;
    float* kbuf  = (float*)(WT + F * F);
    float* qbuf  = kbuf  + (size_t)N * F;
    float* ebuf  = qbuf  + (size_t)N * F;
    float* mbuf  = ebuf  + (size_t)E * NHEADS;
    float* zbuf  = mbuf  + (size_t)N * NHEADS;
    float* hnbuf = zbuf  + (size_t)N * NHEADS;

    // 0) pack weights to column-major bf16 (one-time, tiny)
    gat_pack_w<<<(F * F + 255) / 256, 256, 0, stream>>>(Wk, Wq, W, WkT, WqT, WT);
    // 1) init softmax/accumulator state
    {
        long long tot = (long long)N * NHEADS * F;
        int blocks = (int)((tot + 255) / 256);
        gat_init<<<blocks, 256, 0, stream>>>(mbuf, zbuf, hnbuf, N * NHEADS, tot);
    }
    // 2,3) k = h@Wk, q = h@Wq on WMMA (N % 32 == 0)
    gat_gemm64_wmma<<<N / 32, 64, 0, stream>>>(h, WkT, nullptr, kbuf, N);
    gat_gemm64_wmma<<<N / 32, 64, 0, stream>>>(h, WqT, nullptr, qbuf, N);
    // 4) edge scores + segment max
    gat_edge_score<<<(E * NHEADS + 255) / 256, 256, 0, stream>>>(kbuf, qbuf, src, dst, ebuf, mbuf, E);
    // 5) exp + segment sum
    gat_edge_exp<<<(E * NHEADS + 255) / 256, 256, 0, stream>>>(mbuf, dst, ebuf, zbuf, E);
    // 6) weighted scatter-sum into hn
    gat_aggregate<<<(E * F + 255) / 256, 256, 0, stream>>>(h, ebuf, zbuf, src, dst, hnbuf, E);
    // 7) out = hn @ W + b on WMMA ((N*H) % 32 == 0)
    gat_gemm64_wmma<<<(N * NHEADS) / 32, 64, 0, stream>>>(hnbuf, WT, b, out, N * NHEADS);
}
